// Attention_1494648619198
// MI455X (gfx1250) — matrix operand order repaired and checked by reading, not compiled, not run on hardware
//
#include <hip/hip_runtime.h>
#include <hip/hip_bf16.h>
#include <math.h>

// ---------------- problem constants ----------------
constexpr int BATCH = 2;
constexpr int SEQ   = 2048;
constexpr int DIM   = 2048;
constexpr int NH    = 16;
constexpr int HD    = 128;           // head dim
constexpr float NEGV = -1.0e9f;
constexpr float QSCALE = 0.08838834764831845f;   // 1/sqrt(128)

typedef __bf16 bf16_t;
typedef bf16_t v8bf  __attribute__((ext_vector_type(8)));
typedef bf16_t v16bf __attribute__((ext_vector_type(16)));
typedef float  v8f   __attribute__((ext_vector_type(8)));
typedef unsigned int v4u __attribute__((ext_vector_type(4)));
typedef int v4i __attribute__((ext_vector_type(4)));
typedef int v8i __attribute__((ext_vector_type(8)));

#ifndef __has_builtin
#define __has_builtin(x) 0
#endif
#if __has_builtin(__builtin_amdgcn_tensor_load_to_lds)
#define HAVE_TDM 1
#else
#define HAVE_TDM 0
#endif

constexpr int LROW = 40;   // LDS row stride in elems: 32 data + 4-DW TDM pad

__device__ __forceinline__ v8f zero8() {
    v8f z;
#pragma unroll
    for (int i = 0; i < 8; ++i) z[i] = 0.0f;
    return z;
}

// ---- WMMA fragment loaders (wave32, 16x16x32 bf16; ISA 7.12.2 layouts) ----
// A-matrix 16x32 (MxK): lane<16 -> row=lane, K = kc+0..7 then kc+16..23
//                        lane>=16 -> row=lane-16, K = kc+8..15 then kc+24..31
__device__ __forceinline__ v16bf load_frag_a(const bf16_t* base, int row_stride,
                                             int kc, int lane) {
    const int half = (lane >> 4) & 1;
    const bf16_t* p = base + (size_t)(lane & 15) * row_stride + kc + half * 8;
    union { v16bf v; v8bf h[2]; } u;
    u.h[0] = *(const v8bf*)(p);
    u.h[1] = *(const v8bf*)(p + 16);
    return u.v;
}

// B-matrix 32x16 (KxN), storage is N-major (row = N, contiguous K):
// lane<16 -> col=lane, K = kc+0..7 then kc+16..23 ; lane>=16 -> col=lane-16, K = kc+8..15 then kc+24..31
__device__ __forceinline__ v16bf load_frag_b(const bf16_t* base, int row_stride,
                                             int kc, int lane) {
    const int half = (lane >> 4) & 1;
    const bf16_t* p = base + (size_t)(lane & 15) * row_stride + kc + half * 8;
    union { v16bf v; v8bf h[2]; } u;
    u.h[0] = *(const v8bf*)(p);
    u.h[1] = *(const v8bf*)(p + 16);
    return u.v;
}

__device__ __forceinline__ v8f wmma_bf16(v16bf a, v16bf b, v8f c) {
    return __builtin_amdgcn_wmma_f32_16x16x32_bf16(false, a, false, b,
                                                   (short)0, c, false, false);
}

#if HAVE_TDM
// ---- Tensor Data Mover: DMA a 2D bf16 tile (tile_w x tile_h) from global
// (row stride = `stride` elems) into LDS at byte offset lds_off.
// LDS padding: 4 DWORDs (16B) appended after every 16 DWORDs (64B) -> a
// 32-elem row occupies 80B in LDS (bank-conflict-free fragment reads).
__device__ __forceinline__ void tdm_load_tile_2d(unsigned lds_off,
                                                 const bf16_t* gptr,
                                                 int tensor_w, int tensor_h,
                                                 int tile_w, int tile_h,
                                                 int stride) {
    unsigned long long ga = (unsigned long long)(size_t)gptr;
    v4u g0;
    g0[0] = 1u;                                        // count=1, user mode
    g0[1] = lds_off;                                   // lds_addr
    g0[2] = (unsigned)(ga & 0xFFFFFFFFu);              // global_addr[31:0]
    g0[3] = (unsigned)((ga >> 32) & 0x1FFFFFFu)        // global_addr[56:32]
          | (2u << 30);                                // type = 2 ("image")
    v8i g1;
    g1[0] = (int)((1u << 16)                           // data_size = 2 bytes
                | (1u << 20)                           // pad_enable
                | (3u << 22)                           // pad_interval: 16 DW
                | (3u << 25));                         // pad_amount:   4 DW
    g1[1] = (int)(((unsigned)tensor_w & 0xFFFFu) << 16);          // dim0 lo
    g1[2] = (int)((((unsigned)tensor_w >> 16) & 0xFFFFu)
                | (((unsigned)tensor_h & 0xFFFFu) << 16));        // dim0 hi|dim1 lo
    g1[3] = (int)((((unsigned)tensor_h >> 16) & 0xFFFFu)
                | ((unsigned)tile_w << 16));                      // dim1 hi|tile0
    g1[4] = tile_h;                                    // tile_dim1 (tile_dim2=0)
    g1[5] = stride;                                    // tensor_dim0_stride lo
    g1[6] = 0;
    g1[7] = 0;
    v4i gz = {0, 0, 0, 0};
#if __clang_major__ >= 23
    v8i gz8 = {0, 0, 0, 0, 0, 0, 0, 0};
    __builtin_amdgcn_tensor_load_to_lds(g0, g1, gz, gz, gz8, 0);
#else
    __builtin_amdgcn_tensor_load_to_lds(g0, g1, gz, gz, 0);
#endif
}

__device__ __forceinline__ void wait_tensorcnt0() {
#if __has_builtin(__builtin_amdgcn_s_wait_tensorcnt)
    __builtin_amdgcn_s_wait_tensorcnt(0);
#else
    asm volatile("s_wait_tensorcnt 0x0" ::: "memory");
#endif
}
#endif  // HAVE_TDM

// ---------------- elementwise / layout kernels ----------------
__global__ void cast_f32_to_bf16(const float* __restrict__ in,
                                 bf16_t* __restrict__ out, int n) {
    int i = blockIdx.x * blockDim.x + threadIdx.x;
    if (i < n) out[i] = (bf16_t)in[i];
}

// w: K x N row-major (DIM x DIM) -> wt: N x K row-major (bf16)
__global__ void transpose_cast_bf16(const float* __restrict__ w,
                                    bf16_t* __restrict__ wt, int n) {
    int i = blockIdx.x * blockDim.x + threadIdx.x;
    if (i >= n) return;
    int nrow = i / DIM;
    int k    = i % DIM;
    wt[i] = (bf16_t)w[(size_t)k * DIM + nrow];
}

// t: f32 [B][S][H][HD] -> out: bf16 [B][H][S][HD] with RoPE (+ optional scale)
__global__ void rope_cast_bf16(const float* __restrict__ t,
                               const float* __restrict__ fcos,
                               const float* __restrict__ fsin,
                               bf16_t* __restrict__ out, float scale) {
    int i = blockIdx.x * blockDim.x + threadIdx.x;   // over B*S*H*(HD/2)
    const int HD2 = HD / 2;
    if (i >= BATCH * SEQ * NH * HD2) return;
    int d2 = i % HD2;
    int h  = (i / HD2) % NH;
    int s  = (i / (HD2 * NH)) % SEQ;
    int b  = i / (HD2 * NH * SEQ);
    float c  = fcos[s * HD2 + d2];
    float sn = fsin[s * HD2 + d2];
    const float* p = t + ((size_t)(b * SEQ + s)) * DIM + h * HD + 2 * d2;
    float tr = p[0], ti = p[1];
    float orr = (tr * c - ti * sn) * scale;
    float oii = (tr * sn + ti * c) * scale;
    bf16_t* q = out + ((((size_t)b * NH + h) * SEQ + s) * HD) + 2 * d2;
    q[0] = (bf16_t)orr;
    q[1] = (bf16_t)oii;
}

// v: f32 [B][S][H][HD] -> vt: bf16 [B][H][HD][S]  (transposed per head)
__global__ void v_transpose_bf16(const float* __restrict__ v,
                                 bf16_t* __restrict__ vt) {
    int i = blockIdx.x * blockDim.x + threadIdx.x;   // over B*H*HD*S, s fastest
    if (i >= BATCH * NH * HD * SEQ) return;
    int s = i % SEQ;
    int d = (i / SEQ) % HD;
    int h = (i / (SEQ * HD)) % NH;
    int b = i / (SEQ * HD * NH);
    vt[i] = (bf16_t)v[((size_t)(b * SEQ + s)) * DIM + h * HD + d];
}

// ---------------- GEMM: C(f32 MxN) = A(bf16 MxK) * Bt(bf16 NxK)^T ----------
// 256 threads = 8 waves; block tile 128x128; wave tile 32x64 (2x4 WMMA tiles).
// TDM path: double-buffered 128x32 A/B tiles DMA'd into LDS (padded rows),
// wave 0 drives the Tensor Data Mover, TENSORcnt + barrier gate consumption.
__global__ __launch_bounds__(256) void gemm_bf16_wmma(
        const bf16_t* __restrict__ A, const bf16_t* __restrict__ Bt,
        float* __restrict__ C, int M, int N, int K) {
    const int lane = threadIdx.x & 31;
    const int wid  = threadIdx.x >> 5;          // 0..7
    const int wm   = wid & 3;                   // 4 M-slots
    const int wn   = wid >> 2;                  // 2 N-slots
    const int bm   = blockIdx.x * 128;
    const int bn   = blockIdx.y * 128;
    const int m0   = bm + wm * 32;
    const int n0   = bn + wn * 64;
    const int half = (lane >> 4) & 1;
    const int ln   = lane & 15;

    v8f acc[2][4];
#pragma unroll
    for (int i = 0; i < 2; ++i)
#pragma unroll
        for (int j = 0; j < 4; ++j) acc[i][j] = zero8();

#if HAVE_TDM
    __shared__ __align__(16) bf16_t sa[2][128 * LROW];   // 20 KB
    __shared__ __align__(16) bf16_t sb[2][128 * LROW];   // 20 KB
    const int nk = K / 32;
    if (wid == 0) {
        tdm_load_tile_2d((unsigned)(size_t)&sa[0][0], A + (size_t)bm * K,
                         K, 128, 32, 128, K);
        tdm_load_tile_2d((unsigned)(size_t)&sb[0][0], Bt + (size_t)bn * K,
                         K, 128, 32, 128, K);
    }
    for (int it = 0; it < nk; ++it) {
        if (wid == 0) wait_tensorcnt0();
        __syncthreads();                        // tile it ready; tile it-1 free
        if (wid == 0 && it + 1 < nk) {          // prefetch next tile via TDM
            int k = (it + 1) * 32;
            int nb = (it + 1) & 1;
            tdm_load_tile_2d((unsigned)(size_t)&sa[nb][0],
                             A + (size_t)bm * K + k, K, 128, 32, 128, K);
            tdm_load_tile_2d((unsigned)(size_t)&sb[nb][0],
                             Bt + (size_t)bn * K + k, K, 128, 32, 128, K);
        }
        const bf16_t* at  = &sa[it & 1][0];
        const bf16_t* btl = &sb[it & 1][0];
        // preload all fragments so all ds_load_b128 issue under one wait
        v16bf a0 = load_frag_a(at + (wm * 32) * LROW,      LROW, 0, lane);
        v16bf a1 = load_frag_a(at + (wm * 32 + 16) * LROW, LROW, 0, lane);
        v16bf bb[4];
#pragma unroll
        for (int j = 0; j < 4; ++j)
            bb[j] = load_frag_b(btl + (wn * 64 + j * 16) * LROW, LROW, 0, lane);
#pragma unroll
        for (int j = 0; j < 4; ++j) {
            acc[0][j] = wmma_bf16(a0, bb[j], acc[0][j]);
            acc[1][j] = wmma_bf16(a1, bb[j], acc[1][j]);
        }
    }
#else
    for (int k = 0; k < K; k += 32) {
        v16bf a0 = load_frag_a(A + (size_t)m0 * K, K, k, lane);
        v16bf a1 = load_frag_a(A + (size_t)(m0 + 16) * K, K, k, lane);
#pragma unroll
        for (int j = 0; j < 4; ++j) {
            v16bf b = load_frag_b(Bt + (size_t)(n0 + j * 16) * K, K, k, lane);
            acc[0][j] = wmma_bf16(a0, b, acc[0][j]);
            acc[1][j] = wmma_bf16(a1, b, acc[1][j]);
        }
    }
#endif
    // C/D layout: elem (M = r + half*8 [+16*i], N = j*16 + ln)
#pragma unroll
    for (int i = 0; i < 2; ++i)
#pragma unroll
        for (int j = 0; j < 4; ++j)
#pragma unroll
            for (int r = 0; r < 8; ++r)
                C[(size_t)(m0 + i * 16 + r + half * 8) * N + (n0 + j * 16 + ln)]
                    = acc[i][j][r];
}

// ---------------- flash attention -----------------------------------------
// 128 threads = 4 waves; all 4 waves of a block share one (b,h) and cover 4
// consecutive 16-row q-tiles (SEQ/16 = 128 tiles/head, divisible by 4).
// TDM path: block-shared K/V tiles (32 keys/superblock) DMA'd into LDS,
// double-buffered; uniform superblock count = max wave's causal bound; waves
// guard their own compute (masked keys contribute exp(-1e9-m) ~ 0).
// qbf/kbf: bf16 [B][H][S][HD]; vt: bf16 [B][H][HD][S]; o: f32 [B][S][DIM]
__global__ __launch_bounds__(128) void attn_wmma(
        const bf16_t* __restrict__ qbf, const bf16_t* __restrict__ kbf,
        const bf16_t* __restrict__ vt, float* __restrict__ o) {
    __shared__ __align__(16) bf16_t pshare[4][16 * 32];   // P tile per wave
#if HAVE_TDM
    __shared__ __align__(16) bf16_t sk[2][4][32 * LROW];  // 4 hd-chunks x 32 keys
    __shared__ __align__(16) bf16_t sv[2][128 * LROW];    // 128 hd rows x 32 keys
#endif

    const int lane = threadIdx.x & 31;
    const int wid  = threadIdx.x >> 5;                    // wave in block: 0..3
    const int half = (lane >> 4) & 1;
    const int ln   = lane & 15;

    const int ntile = SEQ / 16;                           // q-tiles per head
    int tile = blockIdx.x * 4 + wid;                      // global q-tile id
    int qt = tile % ntile;
    int h  = (tile / ntile) % NH;
    int b  = tile / (ntile * NH);
    int qbase = qt * 16;

    const bf16_t* qh = qbf + ((size_t)b * NH + h) * SEQ * HD;
    const bf16_t* kh = kbf + ((size_t)b * NH + h) * SEQ * HD;
    const bf16_t* vh = vt  + ((size_t)b * NH + h) * HD * SEQ;

    // Q fragments for all 4 K-chunks of the head dim (reused each key block)
    v16bf aq[4];
#pragma unroll
    for (int kc = 0; kc < 4; ++kc)
        aq[kc] = load_frag_a(qh + (size_t)qbase * HD, HD, kc * 32, lane);

    v8f acc[8];                                           // 16 x 128 output
#pragma unroll
    for (int t = 0; t < 8; ++t) acc[t] = zero8();
    float m[8], l[8];
#pragma unroll
    for (int r = 0; r < 8; ++r) { m[r] = -INFINITY; l[r] = 0.0f; }

    bf16_t* pl = pshare[wid];

#if HAVE_TDM
    const int qt0 = (blockIdx.x * 4) % ntile;             // block-uniform
    const int nsb = (qt0 * 16 + 64) / 32;                 // 32-key superblocks
    if (wid == 0) {
#pragma unroll
        for (int kc = 0; kc < 4; ++kc)
            tdm_load_tile_2d((unsigned)(size_t)&sk[0][kc][0],
                             kh + kc * 32, HD, 32, 32, 32, HD);
        tdm_load_tile_2d((unsigned)(size_t)&sv[0][0],
                         vh, SEQ, 128, 32, 128, SEQ);
    }
    for (int sb = 0; sb < nsb; ++sb) {
        const int kb  = sb * 32;
        const int buf = sb & 1;
        if (wid == 0) wait_tensorcnt0();
        __syncthreads();                        // tiles sb ready; sb-1 free
        if (wid == 0 && sb + 1 < nsb) {         // prefetch next superblock
            const int nk0 = (sb + 1) * 32;
            const int nb  = (sb + 1) & 1;
#pragma unroll
            for (int kc = 0; kc < 4; ++kc)
                tdm_load_tile_2d((unsigned)(size_t)&sk[nb][kc][0],
                                 kh + (size_t)nk0 * HD + kc * 32,
                                 HD, 32, 32, 32, HD);
            tdm_load_tile_2d((unsigned)(size_t)&sv[nb][0],
                             vh + nk0, SEQ, 128, 32, 128, SEQ);
        }
        if (kb >= qbase + 16) continue;         // this wave fully masked here
        // ---- scores from LDS K tiles ----
        v8f s0 = zero8(), s1 = zero8();
#pragma unroll
        for (int kc = 0; kc < 4; ++kc) {
            const bf16_t* skc = &sk[buf][kc][0];
            v16bf b0 = load_frag_b(skc,             LROW, 0, lane);
            v16bf b1 = load_frag_b(skc + 16 * LROW, LROW, 0, lane);
            s0 = wmma_bf16(aq[kc], b0, s0);
            s1 = wmma_bf16(aq[kc], b1, s1);
        }
#else
    for (int kb = 0; kb < qbase + 16; kb += 32) {
        v8f s0 = zero8(), s1 = zero8();
#pragma unroll
        for (int kc = 0; kc < 4; ++kc) {
            v16bf b0 = load_frag_b(kh + (size_t)kb * HD,        HD, kc * 32, lane);
            v16bf b1 = load_frag_b(kh + (size_t)(kb + 16) * HD, HD, kc * 32, lane);
            s0 = wmma_bf16(aq[kc], b0, s0);
            s1 = wmma_bf16(aq[kc], b1, s1);
        }
#endif
        // ---- causal mask + online softmax (rows across 16-lane halves) ----
        const int key0 = kb + ln;
        const int key1 = kb + 16 + ln;
        float pr0[8], pr1[8];
#pragma unroll
        for (int r = 0; r < 8; ++r) {
            int qrow = qbase + r + half * 8;
            float v0 = (key0 <= qrow) ? s0[r] : NEGV;
            float v1 = (key1 <= qrow) ? s1[r] : NEGV;
            float mx = fmaxf(v0, v1);
#pragma unroll
            for (int off = 1; off < 16; off <<= 1)
                mx = fmaxf(mx, __shfl_xor(mx, off, 16));
            float mn = fmaxf(m[r], mx);
            float sc = __expf(m[r] - mn);
            float p0 = __expf(v0 - mn);
            float p1 = __expf(v1 - mn);
            float rs = p0 + p1;
#pragma unroll
            for (int off = 1; off < 16; off <<= 1)
                rs += __shfl_xor(rs, off, 16);
            l[r] = l[r] * sc + rs;
            m[r] = mn;
#pragma unroll
            for (int t = 0; t < 8; ++t) acc[t][r] *= sc;
            pr0[r] = p0;
            pr1[r] = p1;
        }
        // ---- C-layout -> A-layout via LDS (intra-wave) ----
#pragma unroll
        for (int r = 0; r < 8; ++r) {
            int row = r + half * 8;
            pl[row * 32 + ln]      = (bf16_t)pr0[r];
            pl[row * 32 + 16 + ln] = (bf16_t)pr1[r];
        }
        asm volatile("s_wait_dscnt 0" ::: "memory");      // CDNA5 split counter
        union { v16bf v; v8bf hh[2]; } up;
        up.hh[0] = *(const v8bf*)(pl + ln * 32 + half * 8);
        up.hh[1] = *(const v8bf*)(pl + ln * 32 + half * 8 + 16);
        v16bf pf = up.v;
        // ---- O += P (16x32) x V (32x128) ----
#if HAVE_TDM
#pragma unroll
        for (int t = 0; t < 8; ++t) {
            v16bf bv = load_frag_b(&sv[buf][0] + (t * 16) * LROW, LROW, 0, lane);
            acc[t] = wmma_bf16(pf, bv, acc[t]);
        }
    }
#else
#pragma unroll
        for (int t = 0; t < 8; ++t) {
            v16bf bv = load_frag_b(vh + (size_t)(t * 16) * SEQ, SEQ, kb, lane);
            acc[t] = wmma_bf16(pf, bv, acc[t]);
        }
    }
#endif
    // ---- epilogue: normalize, write f32 [B][S][DIM] ----
#pragma unroll
    for (int t = 0; t < 8; ++t) {
#pragma unroll
        for (int r = 0; r < 8; ++r) {
            int qrow = qbase + r + half * 8;
            o[((size_t)(b * SEQ + qrow)) * DIM + h * HD + t * 16 + ln]
                = acc[t][r] / l[r];
        }
    }
}

// ---------------- host orchestration ----------------
extern "C" void kernel_launch(void* const* d_in, const int* in_sizes, int n_in,
                              void* d_out, int out_size, void* d_ws, size_t ws_size,
                              hipStream_t stream) {
    const float* x    = (const float*)d_in[0];
    const float* fcos = (const float*)d_in[1];
    const float* fsin = (const float*)d_in[2];
    // d_in[3] = mask (unused; causal mask computed analytically)
    const float* wq   = (const float*)d_in[4];
    const float* wk   = (const float*)d_in[5];
    const float* wv   = (const float*)d_in[6];
    const float* wo   = (const float*)d_in[7];

    char* ws = (char*)d_ws;
    const size_t SZ_XB  = (size_t)BATCH * SEQ * DIM * sizeof(bf16_t);   // 16 MB
    const size_t SZ_WT  = (size_t)DIM * DIM * sizeof(bf16_t);           //  8 MB
    const size_t SZ_HBF = (size_t)BATCH * NH * SEQ * HD * sizeof(bf16_t); // 16 MB
    bf16_t* XB  = (bf16_t*)(ws);
    bf16_t* WQT = (bf16_t*)(ws + SZ_XB);
    bf16_t* WKT = (bf16_t*)(ws + SZ_XB + SZ_WT);
    bf16_t* WVT = (bf16_t*)(ws + SZ_XB + 2 * SZ_WT);
    bf16_t* WOT = (bf16_t*)(ws + SZ_XB + 3 * SZ_WT);
    bf16_t* QBF = (bf16_t*)(ws + SZ_XB + 4 * SZ_WT);
    bf16_t* KBF = (bf16_t*)(ws + SZ_XB + 4 * SZ_WT + SZ_HBF);
    bf16_t* VT  = (bf16_t*)(ws + SZ_XB + 4 * SZ_WT + 2 * SZ_HBF);
    float*  F32 = (float*)(ws + SZ_XB + 4 * SZ_WT + 3 * SZ_HBF);        // 32 MB
    // total workspace use: 128 MB

    const int nBSD = BATCH * SEQ * DIM;       // 8,388,608
    const int nW   = DIM * DIM;               // 4,194,304
    const int nRope = BATCH * SEQ * NH * (HD / 2);
    const int nVt  = BATCH * NH * HD * SEQ;

    const int T = 256;
    // 1) x -> bf16
    cast_f32_to_bf16<<<(nBSD + T - 1) / T, T, 0, stream>>>(x, XB, nBSD);
    // 2) weights -> transposed bf16 (N x K)
    transpose_cast_bf16<<<(nW + T - 1) / T, T, 0, stream>>>(wq, WQT, nW);
    transpose_cast_bf16<<<(nW + T - 1) / T, T, 0, stream>>>(wk, WKT, nW);
    transpose_cast_bf16<<<(nW + T - 1) / T, T, 0, stream>>>(wv, WVT, nW);
    transpose_cast_bf16<<<(nW + T - 1) / T, T, 0, stream>>>(wo, WOT, nW);

    dim3 ggrid(BATCH * SEQ / 128, DIM / 128);             // (32, 16)
    // 3) Q = x @ wq ; RoPE + scale -> QBF
    gemm_bf16_wmma<<<ggrid, T, 0, stream>>>(XB, WQT, F32, BATCH * SEQ, DIM, DIM);
    rope_cast_bf16<<<(nRope + T - 1) / T, T, 0, stream>>>(F32, fcos, fsin, QBF, QSCALE);
    // 4) K = x @ wk ; RoPE -> KBF
    gemm_bf16_wmma<<<ggrid, T, 0, stream>>>(XB, WKT, F32, BATCH * SEQ, DIM, DIM);
    rope_cast_bf16<<<(nRope + T - 1) / T, T, 0, stream>>>(F32, fcos, fsin, KBF, 1.0f);
    // 5) V = x @ wv ; per-head transpose -> VT
    gemm_bf16_wmma<<<ggrid, T, 0, stream>>>(XB, WVT, F32, BATCH * SEQ, DIM, DIM);
    v_transpose_bf16<<<(nVt + T - 1) / T, T, 0, stream>>>(F32, VT);
    // 6) attention -> F32 (as [B][S][DIM])
    int nqtiles = BATCH * NH * (SEQ / 16);                // 4096 waves
    attn_wmma<<<nqtiles / 4, 128, 0, stream>>>(QBF, KBF, VT, F32);
    // 7) out = attn_out @ wo  (reuse XB for bf16 attn output)
    cast_f32_to_bf16<<<(nBSD + T - 1) / T, T, 0, stream>>>(F32, XB, nBSD);
    gemm_bf16_wmma<<<ggrid, T, 0, stream>>>(XB, WOT, (float*)d_out,
                                            BATCH * SEQ, DIM, DIM);
}
